// Elman_74440373174740
// MI455X (gfx1250) — matrix operand order repaired and checked by reading, not compiled, not run on hardware
//
#include <hip/hip_runtime.h>
#include <hip/hip_bf16.h>

typedef __attribute__((ext_vector_type(16))) __bf16 v16bf;
typedef __attribute__((ext_vector_type(8)))  __bf16 v8bf;
typedef __attribute__((ext_vector_type(8)))  float  v8f;
typedef __attribute__((ext_vector_type(4)))  float  v4f;

#define B_ 64
#define T_ 512
#define I_ 512
#define H_ 1024

// padded LDS row strides (in bf16 elements): +8 elems = +4 dword-banks per row
#define LDSP_I (I_ + 8)   // 520  -> row stride 1040 B (16B-aligned, 4-bank shift/row)
#define LDSP_H (H_ + 8)   // 1032 -> row stride 2064 B (16B-aligned, 4-bank shift/row)

#define SCAN_BLOCKS 32    // 4 m-tiles * 8 n-groups; tiny -> guaranteed co-resident

// ---------------- fp32 -> (bf16 hi, bf16 lo) split ----------------
__global__ void split_bf16_kernel(const float* __restrict__ src,
                                  __bf16* __restrict__ hi,
                                  __bf16* __restrict__ lo, int n) {
    int i = blockIdx.x * blockDim.x + threadIdx.x;
    if (i >= n) return;
    float f = src[i];
    __bf16 h = (__bf16)f;
    hi[i] = h;
    lo[i] = (__bf16)(f - (float)h);
}

// ---------------- phase 1: xp = emb @ Wx^T + bx  (bf16x3 WMMA) ----------------
// Block = 8 waves; all 8 share one 16-row A tile (staged+split once in LDS),
// each wave owns one 16x16 output tile. M=B*T=32768, N=H=1024, K=I=512.
__global__ void __launch_bounds__(256) proj_kernel(const float*  __restrict__ emb,
                                                   const __bf16* __restrict__ WxHi,
                                                   const __bf16* __restrict__ WxLo,
                                                   const float*  __restrict__ bx,
                                                   float*        __restrict__ xp) {
    __shared__ __bf16 aHi[16 * LDSP_I];
    __shared__ __bf16 aLo[16 * LDSP_I];

    const int tid = threadIdx.x;
    const int tm  = blockIdx.x >> 3;        // m-tile (2048)
    const int ng  = (blockIdx.x & 7) * 8;   // first n-tile of this block

    const float* abase = emb + (size_t)tm * 16 * I_;
    for (int v = tid; v < 16 * I_ / 4; v += 256) {      // 2048 float4 chunks
        int row = v >> 7;                                // 128 float4 per row
        int col = (v & 127) * 4;
        v4f f = *(const v4f*)(abase + row * I_ + col);
        int o = row * LDSP_I + col;
#pragma unroll
        for (int j = 0; j < 4; j++) {
            __bf16 h = (__bf16)f[j];
            aHi[o + j] = h;
            aLo[o + j] = (__bf16)(f[j] - (float)h);
        }
    }
    __syncthreads();

    const int lane = tid & 31;
    const int wv   = tid >> 5;
    const int tileN = ng + wv;
    const int hf = lane >> 4;
    const int ln = lane & 15;
    const int nB = tileN * 16 + ln;

    const __bf16* bhrow = WxHi + (size_t)nB * I_;
    const __bf16* blrow = WxLo + (size_t)nB * I_;

    v8f acc = {};
    for (int k0 = 0; k0 < I_; k0 += 32) {
        const int ka = ln * LDSP_I + k0 + hf * 16;
        v8bf a0 = *(const v8bf*)&aHi[ka];
        v8bf a1 = *(const v8bf*)&aHi[ka + 8];
        v8bf l0 = *(const v8bf*)&aLo[ka];
        v8bf l1 = *(const v8bf*)&aLo[ka + 8];
        v16bf ahi, alo;
#pragma unroll
        for (int i = 0; i < 8; i++) {
            ahi[i] = a0[i]; ahi[8 + i] = a1[i];
            alo[i] = l0[i]; alo[8 + i] = l1[i];
        }
        v16bf bhi = *(const v16bf*)(bhrow + k0 + hf * 16);
        v16bf blo = *(const v16bf*)(blrow + k0 + hf * 16);

        acc = __builtin_amdgcn_wmma_f32_16x16x32_bf16(false, ahi, false, bhi, (short)0, acc, false, false);
        acc = __builtin_amdgcn_wmma_f32_16x16x32_bf16(false, ahi, false, blo, (short)0, acc, false, false);
        acc = __builtin_amdgcn_wmma_f32_16x16x32_bf16(false, alo, false, bhi, (short)0, acc, false, false);
    }

    const float bias = bx[nB];
#pragma unroll
    for (int r = 0; r < 8; r++) {
        int m = tm * 16 + r + hf * 8;   // C layout: VGPR r -> row r (+8 for upper lane half)
        xp[(size_t)m * H_ + nB] = acc[r] + bias;
    }
}

// ---------------- phase 2: persistent cooperative scan ----------------
// One launch runs all T=512 steps: h_new = tanh(xp_t + h @ Wh^T + bh).
// 32 co-resident blocks; device-scope release/acquire barrier between steps.
__global__ void __launch_bounds__(256) scan_kernel(__bf16* __restrict__ hA_hi,
                                                   __bf16* __restrict__ hA_lo,
                                                   __bf16* __restrict__ hB_hi,
                                                   __bf16* __restrict__ hB_lo,
                                                   const __bf16* __restrict__ WhHi,
                                                   const __bf16* __restrict__ WhLo,
                                                   const float*  __restrict__ bh,
                                                   float*        __restrict__ out,   // d_out (hidden holds xp)
                                                   unsigned*     __restrict__ counter) {
    __shared__ __bf16 aHi[16 * LDSP_H];
    __shared__ __bf16 aLo[16 * LDSP_H];

    const int tid = threadIdx.x;
    const int tm  = blockIdx.x >> 3;        // m-tile (4)
    const int ng  = (blockIdx.x & 7) * 8;   // first n-tile of this block

    const int lane = tid & 31;
    const int wv   = tid >> 5;
    const int tileN = ng + wv;
    const int hf = lane >> 4;
    const int ln = lane & 15;
    const int nB = tileN * 16 + ln;

    const __bf16* bhrow = WhHi + (size_t)nB * H_;
    const __bf16* blrow = WhLo + (size_t)nB * H_;
    const float   bias  = bh[nB];

    for (int t = 0; t < T_; t++) {
        const __bf16* hInHi = (t & 1) ? hB_hi : hA_hi;
        const __bf16* hInLo = (t & 1) ? hB_lo : hA_lo;
        __bf16* hOutHi = (t & 1) ? hA_hi : hB_hi;
        __bf16* hOutLo = (t & 1) ? hA_lo : hB_lo;

        // stage 16 rows x 1024 bf16 (hi & lo) of h into padded LDS, 16B chunks
        const __bf16* hbase = hInHi + (size_t)tm * 16 * H_;
        const __bf16* lbase = hInLo + (size_t)tm * 16 * H_;
        for (int v = tid; v < 16 * H_ / 8; v += 256) {   // 2048 v8bf chunks per plane
            int row = v >> 7;                             // 128 chunks per row
            int col = (v & 127) * 8;
            int o = row * LDSP_H + col;
            *(v8bf*)&aHi[o] = *(const v8bf*)(hbase + row * H_ + col);
            *(v8bf*)&aLo[o] = *(const v8bf*)(lbase + row * H_ + col);
        }
        __syncthreads();

        v8f acc = {};
#pragma unroll 2
        for (int k0 = 0; k0 < H_; k0 += 32) {
            const int ka = ln * LDSP_H + k0 + hf * 16;
            v8bf a0 = *(const v8bf*)&aHi[ka];
            v8bf a1 = *(const v8bf*)&aHi[ka + 8];
            v8bf l0 = *(const v8bf*)&aLo[ka];
            v8bf l1 = *(const v8bf*)&aLo[ka + 8];
            v16bf ahi, alo;
#pragma unroll
            for (int i = 0; i < 8; i++) {
                ahi[i] = a0[i]; ahi[8 + i] = a1[i];
                alo[i] = l0[i]; alo[8 + i] = l1[i];
            }
            const int kb = k0 + hf * 16;
            v16bf bhi = *(const v16bf*)(bhrow + kb);
            v16bf blo = *(const v16bf*)(blrow + kb);

            acc = __builtin_amdgcn_wmma_f32_16x16x32_bf16(false, ahi, false, bhi, (short)0, acc, false, false);
            acc = __builtin_amdgcn_wmma_f32_16x16x32_bf16(false, ahi, false, blo, (short)0, acc, false, false);
            acc = __builtin_amdgcn_wmma_f32_16x16x32_bf16(false, alo, false, bhi, (short)0, acc, false, false);
        }

#pragma unroll
        for (int r = 0; r < 8; r++) {
            int b = tm * 16 + r + hf * 8;
            size_t hidIdx = (size_t)b * ((size_t)T_ * H_) + (size_t)t * H_ + nB;
            float v  = out[hidIdx] + acc[r] + bias;   // out currently holds xp
            float hv = tanhf(v);
            out[hidIdx] = hv;                          // overwrite xp with hidden state
            __bf16 hh = (__bf16)hv;
            hOutHi[b * H_ + nB] = hh;
            hOutLo[b * H_ + nB] = (__bf16)(hv - (float)hh);
            if (t == T_ - 1) out[(size_t)B_ * T_ * H_ + (size_t)b * H_ + nB] = hv;  // last_h
        }

        // ---- device-scope grid barrier (monotonic counter, deterministic) ----
        __threadfence();        // release h-plane stores to device scope
        __syncthreads();        // all waves of this block done (also guards LDS reuse)
        if (tid == 0) {
            __hip_atomic_fetch_add(counter, 1u, __ATOMIC_ACQ_REL, __HIP_MEMORY_SCOPE_AGENT);
            const unsigned target = (unsigned)SCAN_BLOCKS * (unsigned)(t + 1);
            while (__hip_atomic_load(counter, __ATOMIC_ACQUIRE, __HIP_MEMORY_SCOPE_AGENT) < target)
                __builtin_amdgcn_s_sleep(2);
        }
        __syncthreads();
    }
}

extern "C" void kernel_launch(void* const* d_in, const int* in_sizes, int n_in,
                              void* d_out, int out_size, void* d_ws, size_t ws_size,
                              hipStream_t stream) {
    const float* emb = (const float*)d_in[0];
    const float* h0  = (const float*)d_in[1];
    const float* Wx  = (const float*)d_in[2];
    const float* bx  = (const float*)d_in[3];
    const float* Wh  = (const float*)d_in[4];
    const float* bh  = (const float*)d_in[5];
    float* out = (float*)d_out;

    // Workspace: WhHi 2MB | WhLo 2MB | WxHi 1MB | WxLo 1MB |
    //            hA_hi 128K | hA_lo 128K | hB_hi 128K | hB_lo 128K | counter (64B)
    char* ws = (char*)d_ws;
    __bf16* WhHi = (__bf16*)ws;
    __bf16* WhLo = WhHi + (size_t)H_ * H_;
    __bf16* WxHi = WhLo + (size_t)H_ * H_;
    __bf16* WxLo = WxHi + (size_t)H_ * I_;
    __bf16* hA_hi = WxLo + (size_t)H_ * I_;
    __bf16* hA_lo = hA_hi + (size_t)B_ * H_;
    __bf16* hB_hi = hA_lo + (size_t)B_ * H_;
    __bf16* hB_lo = hB_hi + (size_t)B_ * H_;
    unsigned* counter = (unsigned*)(hB_lo + (size_t)B_ * H_);

    hipMemsetAsync(counter, 0, 64, stream);   // graph-capture legal; reset each call

    split_bf16_kernel<<<(H_ * H_ + 255) / 256, 256, 0, stream>>>(Wh, WhHi, WhLo, H_ * H_);
    split_bf16_kernel<<<(H_ * I_ + 255) / 256, 256, 0, stream>>>(Wx, WxHi, WxLo, H_ * I_);
    split_bf16_kernel<<<(B_ * H_ + 255) / 256, 256, 0, stream>>>(h0, hA_hi, hA_lo, B_ * H_);

    // Phase 1: 2048 m-tiles * 8 n-groups = 16384 blocks, 8 waves each
    proj_kernel<<<16384, 256, 0, stream>>>(emb, WxHi, WxLo, bx, out);

    // Phase 2: single persistent cooperative kernel runs all 512 steps
    scan_kernel<<<SCAN_BLOCKS, 256, 0, stream>>>(hA_hi, hA_lo, hB_hi, hB_lo,
                                                 WhHi, WhLo, bh, out, counter);
}
